// Attention_39556648796173
// MI455X (gfx1250) — matrix operand order repaired and checked by reading, not compiled, not run on hardware
//
#include <hip/hip_runtime.h>
#include <hip/hip_bf16.h>

// ---------------------------------------------------------------------------
// CDNA5 (gfx1250) bf16 WMMA attention layer, wide-fragment edition.
// Fragments are materialized with 2x b128 loads per lane by keeping the
// contraction (K) dimension contiguous for each lane row.
// ---------------------------------------------------------------------------

typedef __attribute__((ext_vector_type(16))) __bf16 v16bf;
typedef __attribute__((ext_vector_type(8)))  __bf16 v8bf;
typedef __attribute__((ext_vector_type(4)))  __bf16 v4bf;
typedef __attribute__((ext_vector_type(8)))  float  v8f;
typedef __attribute__((ext_vector_type(4)))  float  v4f;

__device__ __forceinline__ v16bf frag_cat(v8bf lo, v8bf hi) {
  return __builtin_shufflevector(lo, hi, 0, 1, 2, 3, 4, 5, 6, 7,
                                         8, 9, 10, 11, 12, 13, 14, 15);
}

// A-matrix 16x32 bf16 fragment (ISA 7.12.2): lane&15 = M row, lane>>4 = half.
// k runs: [half*8 .. +7] and [16+half*8 .. +7] -> 2x b128.
// Requires lda % 8 == 0 and 16B-aligned base.
__device__ __forceinline__ v16bf load_a_frag_w(const __bf16* a, int lda) {
  const int lane = threadIdx.x & 31;
  const int half = lane >> 4, m = lane & 15;
  const __bf16* r = a + m * lda + half * 8;
  return frag_cat(*(const v8bf*)r, *(const v8bf*)(r + 16));
}

// B-matrix 32x16 bf16 fragment from an n-major (k-contiguous) tile:
// B(k,n) = bt[n*ldt + k]; lane&15 = n, k runs: [half*8 .. +7] and [16+half*8 .. +7] -> 2x b128.
__device__ __forceinline__ v16bf load_bT_frag(const __bf16* bt, int ldt) {
  const int lane = threadIdx.x & 31;
  const int half = lane >> 4, n = lane & 15;
  const __bf16* r = bt + n * ldt + half * 8;
  return frag_cat(*(const v8bf*)r, *(const v8bf*)(r + 16));
}

__device__ __forceinline__ v8f wmma_bf16(v16bf a, v16bf b, v8f c) {
  return __builtin_amdgcn_wmma_f32_16x16x32_bf16(false, a, false, b,
                                                 (short)0, c, false, false);
}

// ---------------------------------------------------------------------------
// GEMM: C(MxN) = A(MxK)*B(KxN). fp32 in -> bf16 LDS tiles -> WMMA fp32 acc.
// Block tile 128x64, K-step 32, 128 threads = 4 waves.
// Wave owns 32 rows x 64 cols = 8 C fragments (8 WMMA per K-step).
// B staged transposed (BsT[n][k]) so B-fragments are k-contiguous.
// Output type is a template parameter: no runtime branch in the epilogue.
// ---------------------------------------------------------------------------
#define LDA 40   // row stride for As / BsT / VsT / Ps: %8==0 -> 16B rows
template <typename OutT>
__global__ void __launch_bounds__(128)
gemm_bf16(const float* __restrict__ A, const float* __restrict__ B,
          OutT* __restrict__ C, int M, int N, int K) {
  __shared__ __bf16 As[128][LDA];
  __shared__ __bf16 BsT[64][LDA];
  const int tid  = threadIdx.x;
  const int wave = tid >> 5;
  const int lane = tid & 31;
  const int half = lane >> 4, n16 = lane & 15;
  const int m0 = blockIdx.y * 128;
  const int n0 = blockIdx.x * 64;

  v8f acc[2][4] = {};
  for (int k0 = 0; k0 < K; k0 += 32) {
    __syncthreads();
    // Stage A: 128 rows x 32 k = 512 octets, 4 per thread (32B reads, 16B st).
#pragma unroll
    for (int u = 0; u < 4; ++u) {
      const int unit = u * 128 + tid;
      const int r = unit >> 2, oct = unit & 3;
      const v8f f = *(const v8f*)&A[(size_t)(m0 + r) * K + k0 + oct * 8];
      *(v8bf*)&As[r][oct * 8] = __builtin_convertvector(f, v8bf);
    }
    // Stage B transposed: 64 n x 8 k-quads, 4 per thread; reads coalesced
    // across n, 8B packed LDS stores.
#pragma unroll
    for (int u = 0; u < 4; ++u) {
      const int unit = u * 128 + tid;
      const int n = unit & 63, kq = unit >> 6;
      const float* bp = B + (size_t)(k0 + kq * 4) * N + n0 + n;
      v4f t;
#pragma unroll
      for (int j = 0; j < 4; ++j) t[j] = bp[(size_t)j * N];
      *(v4bf*)&BsT[n][kq * 4] = __builtin_convertvector(t, v4bf);
    }
    __syncthreads();
    const v16bf af0 = load_a_frag_w(&As[wave * 32][0], LDA);
    const v16bf af1 = load_a_frag_w(&As[wave * 32 + 16][0], LDA);
#pragma unroll
    for (int c = 0; c < 4; ++c) {
      const v16bf bf = load_bT_frag(&BsT[c * 16][0], LDA);
      acc[0][c] = wmma_bf16(af0, bf, acc[0][c]);
      acc[1][c] = wmma_bf16(af1, bf, acc[1][c]);
    }
  }
  // Epilogue: one row pointer per (a,r); 4 stores at immediate offsets.
#pragma unroll
  for (int a = 0; a < 2; ++a)
#pragma unroll
    for (int r = 0; r < 8; ++r) {
      OutT* cp = C + (size_t)(m0 + wave * 32 + a * 16 + half * 8 + r) * N
                   + n0 + n16;
#pragma unroll
      for (int c = 0; c < 4; ++c) cp[c * 16] = (OutT)acc[a][c][r];
    }
}

// ---------------------------------------------------------------------------
// RoPE (interleaved pairs, theta=1e4) + fp32 -> bf16 pack.
// ---------------------------------------------------------------------------
__global__ void rope_pack(const float* __restrict__ in, const int* __restrict__ pos,
                          __bf16* __restrict__ out, int n_heads, int total) {
  const int idx = blockIdx.x * blockDim.x + threadIdx.x;
  if (idx >= total) return;
  const int i = idx & 63;
  const int h = (idx >> 6) % n_heads;
  const int t = idx / (64 * n_heads);
  const float inv_freq = __powf(10000.0f, -(float)(2 * i) * (1.0f / 128.0f));
  const float ang = (float)pos[t] * inv_freq;
  float s, c;
  __sincosf(ang, &s, &c);
  const size_t base = (size_t)t * (n_heads * 128) + h * 128 + 2 * i;
  const float x0 = in[base], x1 = in[base + 1];
  out[base]     = (__bf16)(x0 * c - x1 * s);
  out[base + 1] = (__bf16)(x0 * s + x1 * c);
}

// ---------------------------------------------------------------------------
// Flash attention: grid (S/64, n_heads, B), 128 threads = 4 waves.
// K tile [key][dim] (dim-contiguous -> QK^T B-frags wide);
// V tile transposed [dim][key] (key-contiguous -> PV B-frags wide).
// Online softmax fp32; P routed through per-wave LDS (k-contiguous rows).
// ---------------------------------------------------------------------------
#define LDK 136
__global__ void __launch_bounds__(128)
attn_fa(const __bf16* __restrict__ Q, const __bf16* __restrict__ Kc,
        const __bf16* __restrict__ Vc, float* __restrict__ O) {
  __shared__ __bf16 Ks[32][LDK];      // [key][dim]
  __shared__ __bf16 VsT[128][LDA];    // [dim][key]
  __shared__ __bf16 Ps[4][16][LDA];   // per-wave P, [m][key]
  const int tid  = threadIdx.x;
  const int wave = tid >> 5;
  const int lane = tid & 31;
  const int half = lane >> 4, n16 = lane & 15;
  const int s0 = blockIdx.x * 64;
  const int h  = blockIdx.y;
  const int b  = blockIdx.z;
  const int kvh = h >> 1;                    // N_REP = 2
  const float scale = 0.08838834764831845f;  // 1/sqrt(128)

  // Per-kernel KV base for this (batch, kv-head): 32-bit offsets below.
  const __bf16* kbase = Kc + (size_t)b * 1024 * 1024 + kvh * 128;
  const __bf16* vbase = Vc + (size_t)b * 1024 * 1024 + kvh * 128;

  // Q fragments: 4 chunks of K=32 over head_dim 128 (2x global b128 each).
  const __bf16* qbase =
      Q + (size_t)(b * 1024 + s0 + wave * 16) * 2048 + h * 128;
  v16bf qf[4];
#pragma unroll
  for (int f = 0; f < 4; ++f) qf[f] = load_a_frag_w(qbase + f * 32, 2048);

  v8f o[8] = {};
  float m_i[8], l_i[8];
#pragma unroll
  for (int r = 0; r < 8; ++r) { m_i[r] = -INFINITY; l_i[r] = 0.f; }

  const int ntiles = s0 / 32 + 2;            // causal: keys <= s0+63
  for (int kt = 0; kt < ntiles; ++kt) {
    const int sk0 = kt * 32;
    __syncthreads();
    // Stage K [key][dim]: 512 octets, 4 per thread, 16B loads+stores.
#pragma unroll
    for (int u = 0; u < 4; ++u) {
      const int unit = u * 128 + tid;
      const int kr = unit >> 4, oct = unit & 15;
      *(v8bf*)&Ks[kr][oct * 8] =
          *(const v8bf*)&kbase[(sk0 + kr) * 1024 + oct * 8];
    }
    // Stage V transposed [dim][key]: 128 d x 8 key-quads, 8 per thread.
#pragma unroll
    for (int u = 0; u < 8; ++u) {
      const int unit = u * 128 + tid;
      const int d = unit & 127, kq = unit >> 7;
      v4bf t;
#pragma unroll
      for (int j = 0; j < 4; ++j) t[j] = vbase[(sk0 + kq * 4 + j) * 1024 + d];
      *(v4bf*)&VsT[d][kq * 4] = t;
    }
    __syncthreads();

    // --- S = Q K^T : two 16x16 C fragments per wave ---
    float sv[2][8];
#pragma unroll
    for (int c = 0; c < 2; ++c) {
      v8f sc = {};
#pragma unroll
      for (int f = 0; f < 4; ++f) {
        const v16bf bfr = load_bT_frag(&Ks[c * 16][f * 32], LDK);
        sc = wmma_bf16(qf[f], bfr, sc);
      }
#pragma unroll
      for (int r = 0; r < 8; ++r) {
        const int srow = s0 + wave * 16 + half * 8 + r;
        const int key  = sk0 + c * 16 + n16;
        const float v  = sc[r] * scale;
        sv[c][r] = (srow < key) ? -INFINITY : v;   // causal mask
      }
    }

    // --- online softmax (row reductions across 16 lanes of each half) ---
#pragma unroll
    for (int r = 0; r < 8; ++r) {
      float t = fmaxf(sv[0][r], sv[1][r]);
#pragma unroll
      for (int mk = 1; mk < 16; mk <<= 1) t = fmaxf(t, __shfl_xor(t, mk, 32));
      const float mn   = fmaxf(m_i[r], t);
      const float corr = __expf(m_i[r] - mn);
      m_i[r] = mn;
      const float p0 = __expf(sv[0][r] - mn);
      const float p1 = __expf(sv[1][r] - mn);
      Ps[wave][half * 8 + r][n16]      = (__bf16)p0;
      Ps[wave][half * 8 + r][16 + n16] = (__bf16)p1;
      float s = p0 + p1;
#pragma unroll
      for (int mk = 1; mk < 16; mk <<= 1) s += __shfl_xor(s, mk, 32);
      l_i[r] = l_i[r] * corr + s;
#pragma unroll
      for (int d = 0; d < 8; ++d) o[d][r] *= corr;
    }
    __syncthreads();

    // --- O += P * V ---
    const v16bf pa = load_a_frag_w(&Ps[wave][0][0], LDA);
#pragma unroll
    for (int d = 0; d < 8; ++d) {
      const v16bf bfr = load_bT_frag(&VsT[d * 16][0], LDA);
      o[d] = wmma_bf16(pa, bfr, o[d]);
    }
  }

  // --- normalize + store (fp32, [T, n_heads*128] head-major) ---
#pragma unroll
  for (int r = 0; r < 8; ++r) {
    const float inv = 1.0f / l_i[r];
    float* op = O + (size_t)(b * 1024 + s0 + wave * 16 + half * 8 + r) * 2048
                  + h * 128 + n16;
#pragma unroll
    for (int d = 0; d < 8; ++d) op[d * 16] = o[d][r] * inv;
  }
}

// ---------------------------------------------------------------------------
// Host launch. Workspace layout (bytes):
//   [0,   64Mi) q_f32  (reused as attn_f32 after RoPE consumes it)
//   [64,  96Mi) k_f32
//   [96, 128Mi) q_bf16
//   [128,144Mi) k_bf16
//   [144,160Mi) v_bf16
// ---------------------------------------------------------------------------
extern "C" void kernel_launch(void* const* d_in, const int* in_sizes, int n_in,
                              void* d_out, int out_size, void* d_ws, size_t ws_size,
                              hipStream_t stream) {
  (void)in_sizes; (void)n_in; (void)out_size; (void)ws_size;
  const float* x  = (const float*)d_in[0];
  const float* wq = (const float*)d_in[1];
  const float* wk = (const float*)d_in[2];
  const float* wv = (const float*)d_in[3];
  const float* wo = (const float*)d_in[4];
  const int* positions = (const int*)d_in[7];
  float* out = (float*)d_out;

  char* ws = (char*)d_ws;
  float*  qf   = (float*)(ws + 0);
  float*  kf   = (float*)(ws + (64ull  << 20));
  __bf16* qb   = (__bf16*)(ws + (96ull  << 20));
  __bf16* kb   = (__bf16*)(ws + (128ull << 20));
  __bf16* vb   = (__bf16*)(ws + (144ull << 20));
  float*  attn = qf;   // q_f32 dead after RoPE -> reuse for attention output

  const dim3 blk(128);
  gemm_bf16<float> <<<dim3(2048 / 64, 8192 / 128), blk, 0, stream>>>(x, wq, qf, 8192, 2048, 2048);
  gemm_bf16<float> <<<dim3(1024 / 64, 8192 / 128), blk, 0, stream>>>(x, wk, kf, 8192, 1024, 2048);
  gemm_bf16<__bf16><<<dim3(1024 / 64, 8192 / 128), blk, 0, stream>>>(x, wv, vb, 8192, 1024, 2048);

  const int tq = 8192 * 16 * 64;
  const int tk = 8192 * 8 * 64;
  rope_pack<<<(tq + 255) / 256, 256, 0, stream>>>(qf, positions, qb, 16, tq);
  rope_pack<<<(tk + 255) / 256, 256, 0, stream>>>(kf, positions, kb, 8, tk);

  attn_fa<<<dim3(16, 16, 8), blk, 0, stream>>>(qb, kb, vb, attn);

  gemm_bf16<float><<<dim3(2048 / 64, 8192 / 128), blk, 0, stream>>>(attn, wo, out, 8192, 2048, 2048);
}
